// GCNModelVAE_11175504904298
// MI455X (gfx1250) — matrix operand order repaired and checked by reading, not compiled, not run on hardware
//
#include <hip/hip_runtime.h>
#include <hip/hip_bf16.h>

#define N_DRUG 11000
#define N_PRO  1000
#define NTOT   12000
#define EMB    128
#define H1d    256
#define H2d    128
#define NEDGE  (NTOT * 16)
#define CONV_T 121
#define FCK    3872   // 32*121

typedef __attribute__((ext_vector_type(16))) __bf16 bf16x16;
typedef __attribute__((ext_vector_type(2)))  __bf16 bf16x2;
typedef __attribute__((ext_vector_type(8)))  float  floatx8;
// 4-float vector with only 4-byte alignment guarantee (lda may be odd, e.g. 167)
typedef float float4u __attribute__((ext_vector_type(4), aligned(4)));

__device__ __forceinline__ __bf16 f2bf(float f) { return static_cast<__bf16>(f); }

// ---------------------------------------------------------------------------
// Tiled GEMM: C[M,N] = act(A[M,K] @ B[K,N] + bias), f32 in/out, bf16 WMMA
// 16x16x32 with f32 accumulation. Each wave owns a 16 x (16*NT) strip of C:
// the A fragment is loaded once per k-step and reused across NT WMMAs.
// All loads branch-free (indices clamped, stores guarded).
// ---------------------------------------------------------------------------
template <int TRANSB, int NT>
__global__ void __launch_bounds__(256)
gemm_bf16_wmma(const float* __restrict__ A, int lda,
               const float* __restrict__ B, int ldb,
               const float* __restrict__ bias,
               float* __restrict__ C, int ldc,
               int M, int N, int K, int act)
{
  const int lane = threadIdx.x & 31;
  const int wave = threadIdx.x >> 5;
  const int tm   = blockIdx.y;
  const int ntN  = (N + 15) >> 4;
  const int tn0  = (blockIdx.x * 8 + wave) * NT;
  if (tn0 >= ntN) return;                       // wave-uniform exit

  const int row   = lane & 15;
  const int khalf = lane >> 4;

  const int am  = tm * 16 + row;
  const int amc = am < M ? am : M - 1;          // clamp: OOB rows compute garbage
  const float* Ar = A + (size_t)amc * lda;

  int bcol[NT], bcolc[NT];
  const float* Bc[NT];
  #pragma unroll
  for (int jt = 0; jt < NT; ++jt) {
    bcol[jt]  = (tn0 + jt) * 16 + row;
    bcolc[jt] = bcol[jt] < N ? bcol[jt] : N - 1;
    Bc[jt]    = B + (TRANSB ? (size_t)bcolc[jt] * ldb : (size_t)bcolc[jt]);
  }

  floatx8 acc[NT];
  #pragma unroll
  for (int jt = 0; jt < NT; ++jt)
    acc[jt] = (floatx8){0.f, 0.f, 0.f, 0.f, 0.f, 0.f, 0.f, 0.f};

  const int kfull = K & ~31;

  for (int k0 = 0; k0 < kfull; k0 += 32) {
    __builtin_prefetch(Ar + k0 + 128, 0, 3);
    // A fragment: two contiguous 8-float runs per lane -> 4x b128, loaded once
    bf16x16 a;
    {
      const float4u* p0 = (const float4u*)(Ar + k0 + khalf * 16);
      const float4u* p1 = (const float4u*)(Ar + k0 + khalf * 16 + 8);
      float4u r0 = p0[0], r1 = p0[1], r2 = p1[0], r3 = p1[1];
      #pragma unroll
      for (int j = 0; j < 4; ++j) {
        a[j]      = f2bf(r0[j]);
        a[4 + j]  = f2bf(r1[j]);
        a[8 + j]  = f2bf(r2[j]);
        a[12 + j] = f2bf(r3[j]);
      }
    }
    #pragma unroll
    for (int jt = 0; jt < NT; ++jt) {
      bf16x16 b;
      if (TRANSB) {
        const float4u* bp = (const float4u*)(Bc[jt] + k0 + khalf * 16);
        float4u s0 = bp[0], s1 = bp[1], s2 = bp[2], s3 = bp[3];
        #pragma unroll
        for (int j = 0; j < 4; ++j) {
          b[j]      = f2bf(s0[j]);
          b[4 + j]  = f2bf(s1[j]);
          b[8 + j]  = f2bf(s2[j]);
          b[12 + j] = f2bf(s3[j]);
        }
      } else {
        // column bcolc, 16 consecutive K rows; incremental pointer
        const float* bp = Bc[jt] + (size_t)(k0 + khalf * 16) * ldb;
        float t[16];
        #pragma unroll
        for (int j = 0; j < 16; ++j) { t[j] = *bp; bp += ldb; }
        #pragma unroll
        for (int j = 0; j < 16; ++j) b[j] = f2bf(t[j]);
      }
      acc[jt] = __builtin_amdgcn_wmma_f32_16x16x32_bf16(false, a, false, b,
                                                        (short)0, acc[jt], false, false);
    }
  }

  if (kfull < K) {   // ragged K tail (e.g. K=167): clamped loads + cndmask
    bf16x16 a;
    #pragma unroll
    for (int j = 0; j < 16; ++j) {
      int ka  = kfull + khalf * 16 + j;
      int kac = ka < K ? ka : K - 1;
      float fa = Ar[kac];
      a[j] = f2bf(ka < K ? fa : 0.f);
    }
    #pragma unroll
    for (int jt = 0; jt < NT; ++jt) {
      bf16x16 b;
      #pragma unroll
      for (int j = 0; j < 16; ++j) {
        int kb  = kfull + khalf * 16 + j;
        int kbc = kb < K ? kb : K - 1;
        float fb = TRANSB ? Bc[jt][kbc] : Bc[jt][(size_t)kbc * ldb];
        b[j] = f2bf(kb < K ? fb : 0.f);
      }
      acc[jt] = __builtin_amdgcn_wmma_f32_16x16x32_bf16(false, a, false, b,
                                                        (short)0, acc[jt], false, false);
    }
  }

  // store: C VGPR r -> M = r + khalf*8, N = bcol[jt]
  #pragma unroll
  for (int jt = 0; jt < NT; ++jt) {
    const bool colOK = bcol[jt] < N;
    float bv = bias ? bias[bcolc[jt]] : 0.f;
    #pragma unroll
    for (int r = 0; r < 8; ++r) {
      int m = tm * 16 + r + khalf * 8;
      if (m < M && colOK) {
        float v = acc[jt][r] + bv;
        if (act) v = v > 0.f ? v : 0.f;
        C[(size_t)m * ldc + bcol[jt]] = v;
      }
    }
  }
}

// ---------------------------------------------------------------------------
// Protein conv1d as 8 shifted GEMMs per protein:
//   c[o,t] = sum_k ( W_k[32x1024] @ E_p[1024x128] )[o, t+k]
// E_p staged in LDS transposed [128 emb][256 i] bf16, row stride 258
// (129 dwords -> conflict-free lane->bank mapping). Each of 8 waves owns a
// 16x16 tile and keeps 8 shift accumulators live in VGPRs; shifts merged via
// ds_add_f32 into a 32x128 LDS accumulator.
// ---------------------------------------------------------------------------
#define ET_STRIDE 258

__global__ void __launch_bounds__(256)
prot_conv_wmma(const int* __restrict__ pro_x, const float* __restrict__ emb_xt,
               const float* __restrict__ conv_w, const float* __restrict__ conv_b,
               float* __restrict__ c_out /* [N_PRO][FCK] */)
{
  extern __shared__ char smem[];
  __bf16* Et  = (__bf16*)smem;                              // [128][258] bf16
  float*  accl = (float*)(smem + 128 * ET_STRIDE * 2);      // [32][128]  f32

  const int p    = blockIdx.x;
  const int tid  = threadIdx.x;
  const int lane = tid & 31;
  const int wave = tid >> 5;
  const int tmw  = wave >> 3;      // always 0 with 8 waves : o-tile
  const int tnw  = wave & 7;       // 0..7 : e-tile
  const int row   = lane & 15;
  const int khalf = lane >> 4;
  const int o     = tmw * 16 + row;              // A row (output channel)
  const int col   = tnw * 16 + row;              // B col (embedding pos)

  for (int i = tid; i < 32 * 128; i += 256) accl[i] = 0.f;

  floatx8 ck[8];
  #pragma unroll
  for (int k = 0; k < 8; ++k) ck[k] = (floatx8){0.f,0.f,0.f,0.f,0.f,0.f,0.f,0.f};

  for (int i0 = 0; i0 < 1024; i0 += 256) {
    __syncthreads();
    // stage chunk transposed: Et[e][r] = emb_xt[pro_x[p, i0+r], e]; branch-free
    for (int idx = tid; idx < 256 * 128; idx += 256) {
      int r = idx & 255, e = idx >> 8;
      int i = i0 + r;
      int ic = i < 1000 ? i : 999;
      int s  = pro_x[p * 1000 + ic];
      float v = emb_xt[(size_t)s * EMB + e];
      Et[e * ET_STRIDE + r] = f2bf(i < 1000 ? v : 0.f);
    }
    __syncthreads();

    for (int kk0 = 0; kk0 < 256; kk0 += 32) {
      // B fragment: pairs (kk, kk+1) are contiguous in Et -> one b32 LDS load
      bf16x16 bf;
      const __bf16* ecol = Et + col * ET_STRIDE + kk0 + khalf * 16;
      #pragma unroll
      for (int v = 0; v < 8; ++v) {
        bf16x2 t = *(const bf16x2*)(ecol + 2 * v);
        bf[2 * v]     = t[0];
        bf[2 * v + 1] = t[1];
      }
      // A fragments for all 8 kernel shifts: conv_w[o, i, 0..7] = 2x float4
      bf16x16 af[8];
      #pragma unroll
      for (int half = 0; half < 2; ++half) {
        #pragma unroll
        for (int j = 0; j < 8; ++j) {
          int i  = i0 + kk0 + khalf * 16 + half * 8 + j;
          int ic = i < 1000 ? i : 999;
          const float4u* wr = (const float4u*)(conv_w + ((size_t)o * 1000 + ic) * 8);
          float4u w0 = wr[0], w1 = wr[1];
          bool ok = i < 1000;
          #pragma unroll
          for (int k = 0; k < 4; ++k) {
            af[k][half * 8 + j]     = f2bf(ok ? w0[k] : 0.f);
            af[4 + k][half * 8 + j] = f2bf(ok ? w1[k] : 0.f);
          }
        }
      }
      #pragma unroll
      for (int k = 0; k < 8; ++k)
        ck[k] = __builtin_amdgcn_wmma_f32_16x16x32_bf16(false, af[k], false, bf,
                                                        (short)0, ck[k], false, false);
    }
  }
  __syncthreads();

  // merge shifts: acc[o][e-k] += Mk[o][e]   (ds_add_f32)
  #pragma unroll
  for (int k = 0; k < 8; ++k) {
    #pragma unroll
    for (int r = 0; r < 8; ++r) {
      int oo = tmw * 16 + r + khalf * 8;
      int t  = col - k;
      if (t >= 0 && t < CONV_T) atomicAdd(&accl[oo * 128 + t], ck[k][r]);
    }
  }
  __syncthreads();

  for (int idx = tid; idx < 32 * CONV_T; idx += 256) {
    int oo = idx / CONV_T;
    c_out[(size_t)p * FCK + idx] = accl[oo * 128 + (idx - oo * CONV_T)] + conv_b[oo];
  }
}

// ---------------------------------------------------------------------------
// COO spmm: Y[rows[e], :] += vals[e] * X[cols[e], :]   (Y pre-zeroed)
// ---------------------------------------------------------------------------
__global__ void spmm_atomic(const int* __restrict__ rows, const int* __restrict__ cols,
                            const float* __restrict__ vals,
                            const float* __restrict__ X, float* __restrict__ Y,
                            int nEdges, int F)
{
  int idx = blockIdx.x * blockDim.x + threadIdx.x;
  int chunks = F >> 2;
  if (idx >= nEdges * chunks) return;
  int e = idx / chunks;
  int c = (idx - e * chunks) << 2;
  float v = vals[e];
  const float4 s = *(const float4*)(X + (size_t)cols[e] * F + c);
  float* dst = Y + (size_t)rows[e] * F + c;
  atomicAdd(dst + 0, v * s.x);
  atomicAdd(dst + 1, v * s.y);
  atomicAdd(dst + 2, v * s.z);
  atomicAdd(dst + 3, v * s.w);
}

__global__ void fill_zero(float* p, int n) {
  int i = blockIdx.x * blockDim.x + threadIdx.x;
  if (i < n) p[i] = 0.f;
}

__global__ void relu_inplace(float* p, int n) {
  int i = blockIdx.x * blockDim.x + threadIdx.x;
  if (i < n) { float v = p[i]; p[i] = v > 0.f ? v : 0.f; }
}

// ---------------------------------------------------------------------------
static inline void launch_gemm(const float* A, int lda, const float* B, int ldb,
                               int transB, const float* bias, float* C, int ldc,
                               int M, int N, int K, int act, hipStream_t s)
{
  int tM = (M + 15) / 16, tN = (N + 15) / 16;
  if (transB) {   // large-N path (z z^T): 4 N-tiles per wave
    dim3 grid((tN + 31) / 32, tM);
    gemm_bf16_wmma<1, 4><<<grid, 256, 0, s>>>(A, lda, B, ldb, bias, C, ldc, M, N, K, act);
  } else {        // small-N weight GEMMs: 1 N-tile per wave
    dim3 grid((tN + 7) / 8, tM);
    gemm_bf16_wmma<0, 1><<<grid, 256, 0, s>>>(A, lda, B, ldb, bias, C, ldc, M, N, K, act);
  }
}

extern "C" void kernel_launch(void* const* d_in, const int* in_sizes, int n_in,
                              void* d_out, int out_size, void* d_ws, size_t ws_size,
                              hipStream_t stream) {
  const float* drug_x   = (const float*)d_in[0];
  const int*   pro_x    = (const int*)  d_in[1];
  const int*   adj_rows = (const int*)  d_in[2];
  const int*   adj_cols = (const int*)  d_in[3];
  const float* adj_vals = (const float*)d_in[4];
  const float* w1 = (const float*)d_in[5];   const float* b1 = (const float*)d_in[6];
  const float* w2 = (const float*)d_in[7];   const float* b2 = (const float*)d_in[8];
  const float* w3 = (const float*)d_in[9];   const float* b3 = (const float*)d_in[10];
  const float* emb_xt = (const float*)d_in[11];
  const float* conv_w = (const float*)d_in[12];
  const float* conv_b = (const float*)d_in[13];
  const float* fc_w   = (const float*)d_in[14];
  const float* fc_b   = (const float*)d_in[15];
  const float* gc1_w  = (const float*)d_in[16];
  const float* gc2_w  = (const float*)d_in[17];
  const float* gc3_w  = (const float*)d_in[18];

  float* ws = (float*)d_ws;
  float* x    = ws;                    // [12000][128]            1,536,000
  float* bufA = ws + 1536000;          // h / c_ws / xw1   (max) 3,872,000
  float* bufB = ws + 5408000;          // h2 / h1          (max) 3,072,000
  float* bufC = ws + 8480000;          // hw                     1,536,000

  float* adj_rec = (float*)d_out;                       // [12000][12000]
  float* mu      = adj_rec + (size_t)NTOT * NTOT;       // [12000][128]
  float* logvar  = mu + (size_t)NTOT * H2d;             // [12000][128]

  // ---- drug MLP ----
  launch_gemm(drug_x, 167, w1, EMB, 0, b1, bufA, EMB, N_DRUG, EMB, 167, 1, stream);
  launch_gemm(bufA, EMB, w2, 2 * EMB, 0, b2, bufB, 2 * EMB, N_DRUG, 2 * EMB, EMB, 1, stream);
  launch_gemm(bufB, 2 * EMB, w3, EMB, 0, b3, x, EMB, N_DRUG, EMB, 2 * EMB, 1, stream);

  // ---- protein conv (WMMA, LDS-staged) + fc ----
  size_t conv_lds = 128 * ET_STRIDE * 2 + 32 * 128 * 4;   // ~82 KB of 320 KB WGP LDS
  (void)hipFuncSetAttribute((const void*)prot_conv_wmma,
                            hipFuncAttributeMaxDynamicSharedMemorySize, (int)conv_lds);
  prot_conv_wmma<<<N_PRO, 256, conv_lds, stream>>>(pro_x, emb_xt, conv_w, conv_b, bufA);
  launch_gemm(bufA, FCK, fc_w, EMB, 0, fc_b, x + (size_t)N_DRUG * EMB, EMB,
              N_PRO, EMB, FCK, 0, stream);

  // ---- GCN layer 1: h1 = relu(spmm(adj, x @ gc1_w)) ----
  launch_gemm(x, EMB, gc1_w, H1d, 0, nullptr, bufA, H1d, NTOT, H1d, EMB, 0, stream);
  {
    int n = NTOT * H1d;
    fill_zero<<<(n + 255) / 256, 256, 0, stream>>>(bufB, n);
    int total = NEDGE * (H1d / 4);
    spmm_atomic<<<(total + 255) / 256, 256, 0, stream>>>(adj_rows, adj_cols, adj_vals,
                                                         bufA, bufB, NEDGE, H1d);
    relu_inplace<<<(n + 255) / 256, 256, 0, stream>>>(bufB, n);
  }

  // ---- mu = spmm(adj, h1 @ gc2_w) ----
  launch_gemm(bufB, H1d, gc2_w, H2d, 0, nullptr, bufC, H2d, NTOT, H2d, H1d, 0, stream);
  {
    int n = NTOT * H2d;
    fill_zero<<<(n + 255) / 256, 256, 0, stream>>>(mu, n);
    int total = NEDGE * (H2d / 4);
    spmm_atomic<<<(total + 255) / 256, 256, 0, stream>>>(adj_rows, adj_cols, adj_vals,
                                                         bufC, mu, NEDGE, H2d);
  }

  // ---- logvar = spmm(adj, h1 @ gc3_w) ----
  launch_gemm(bufB, H1d, gc3_w, H2d, 0, nullptr, bufC, H2d, NTOT, H2d, H1d, 0, stream);
  {
    int n = NTOT * H2d;
    fill_zero<<<(n + 255) / 256, 256, 0, stream>>>(logvar, n);
    int total = NEDGE * (H2d / 4);
    spmm_atomic<<<(total + 255) / 256, 256, 0, stream>>>(adj_rows, adj_cols, adj_vals,
                                                         bufC, logvar, NEDGE, H2d);
  }

  // ---- adj_rec = mu @ mu^T  (37 GFLOP, 576 MB out ~ 25 us at 23.3 TB/s) ----
  launch_gemm(mu, H2d, mu, H2d, 1, nullptr, adj_rec, NTOT, NTOT, NTOT, H2d, 0, stream);
}
